// BitNetAttention_90709709291804
// MI455X (gfx1250) — matrix operand order repaired and checked by reading, not compiled, not run on hardware
//
#include <hip/hip_runtime.h>
#include <hip/hip_bf16.h>

// ---------------------------------------------------------------------------
// BitNet attention for MI455X (gfx1250, wave32, WMMA 16x16x32 f16/f32-acc)
// GEMMs use double-buffered GLOBAL_LOAD_ASYNC_TO_LDS_B128 staging (ASYNCcnt).
// ---------------------------------------------------------------------------

typedef __attribute__((ext_vector_type(16))) _Float16 v16h;
typedef __attribute__((ext_vector_type(8)))  float    v8f;

namespace bn {
constexpr int B  = 2;
constexpr int S  = 2048;
constexpr int D  = 2048;
constexpr int H  = 16;
constexpr int HD = 128;          // head dim
constexpr int M  = B * S;        // 4096 GEMM rows
constexpr float RSCALE = 0.088388347648318447f;   // 1/sqrt(128)
constexpr float THRESH = 2.0f / 3.0f;

// ws layout (bytes)
constexpr size_t DD      = (size_t)D * D;          // 4,194,304
constexpr size_t BSD     = (size_t)B * S * D;      // 8,388,608
constexpr size_t OFS_SUM = 0;                      // 4 floats
constexpr size_t OFS_W   = 256;                    // 4 x DD f16
constexpr size_t OFS_X16 = OFS_W   + 4 * DD * 2;   // BSD f16 (also att16 later)
constexpr size_t OFS_Q   = OFS_X16 + BSD * 2;      // BSD f16, [B,H,S,HD]
constexpr size_t OFS_K   = OFS_Q   + BSD * 2;      // BSD f16, [B,H,S,HD]
constexpr size_t OFS_VT  = OFS_K   + BSD * 2;      // BSD f16, [B,H,HD,S]
constexpr size_t OFS_M   = OFS_VT  + BSD * 2;      // B*H*S f32 row-max
constexpr size_t OFS_L   = OFS_M   + (size_t)B * H * S * 4; // B*H*S f32 sumexp
} // namespace bn

// --- WMMA fragment loaders (ISA 7.12.2 layouts) ----------------------------
// A (16x32 f16): lane L -> row m=L&15; lanes 0-15 hold K = {k0..k0+7, k0+16..k0+23},
//                lanes 16-31 hold K = {k0+8..k0+15, k0+24..k0+31}.
__device__ __forceinline__ v16h load_frag_a(const _Float16* p0, int ld, int k0, int lane) {
  const int m = lane & 15, hf = lane >> 4;
  const _Float16* p = p0 + (size_t)m * ld + k0 + hf * 8;
  union { v16h v; float4 f[2]; } u;
  u.f[0] = *(const float4*)(p);
  u.f[1] = *(const float4*)(p + 16);
  return u.v;
}
// B (32x16 f16) from column-major storage (column n contiguous over K):
// lane L -> col n=L&15; lanes 0-15 hold K = {k0..k0+7, k0+16..k0+23},
// lanes 16-31 hold K = {k0+8..k0+15, k0+24..k0+31}.
__device__ __forceinline__ v16h load_frag_b(const _Float16* p0, int ld, int k0, int lane) {
  const int n = lane & 15, hf = lane >> 4;
  const _Float16* p = p0 + (size_t)n * ld + k0 + hf * 8;
  union { v16h v; float4 f[2]; } u;
  u.f[0] = *(const float4*)(p);
  u.f[1] = *(const float4*)(p + 16);
  return u.v;
}
#define WMMA_F16(a, b, c) \
  __builtin_amdgcn_wmma_f32_16x16x32_f16(false, (a), false, (b), (short)0, (c), false, false)

// 16-byte async DMA: global -> LDS (tracked with ASYNCcnt, not LOADcnt)
__device__ __forceinline__ void async_copy16(const _Float16* gptr, _Float16* lptr) {
  const unsigned lds_off = (unsigned)(uintptr_t)lptr;  // generic LDS addr: low 32 = offset
  asm volatile("global_load_async_to_lds_b128 %0, %1, off"
               :: "v"(lds_off), "v"(gptr) : "memory");
}

// --- Stage 1: sum(|w|) ------------------------------------------------------
__global__ __launch_bounds__(256) void abs_sum_kernel(const float* __restrict__ w,
                                                      float* __restrict__ out, int n) {
  __shared__ float red[256];
  float s = 0.f;
  for (int i = blockIdx.x * blockDim.x + threadIdx.x; i < n; i += gridDim.x * blockDim.x)
    s += fabsf(w[i]);
  red[threadIdx.x] = s;
  __syncthreads();
  for (int off = 128; off > 0; off >>= 1) {
    if ((int)threadIdx.x < off) red[threadIdx.x] += red[threadIdx.x + off];
    __syncthreads();
  }
  if (threadIdx.x == 0) atomicAdd(out, red[0]);
}

// --- Stage 2: ternary quantize to f16 (scale factored out) ------------------
__global__ __launch_bounds__(256) void quantize_kernel(const float* __restrict__ w,
                                                       const float* __restrict__ sumabs,
                                                       _Float16* __restrict__ wq, int n) {
  const float scale = fminf(fmaxf(sumabs[0] / (float)n, 1e-5f), 1000.0f);
  const float inv = 1.0f / scale;
  int i = blockIdx.x * blockDim.x + threadIdx.x;
  if (i < n) {
    float wn = w[i] * inv;
    float q = (wn > bn::THRESH) ? 1.0f : ((wn < -bn::THRESH) ? -1.0f : 0.0f);
    wq[i] = (_Float16)q;
  }
}

// --- Stage 3: fp32 -> f16 ---------------------------------------------------
__global__ __launch_bounds__(256) void convert_kernel(const float* __restrict__ x,
                                                      _Float16* __restrict__ y, int n) {
  int i = blockIdx.x * blockDim.x + threadIdx.x;
  if (i < n) y[i] = (_Float16)x[i];
}

// --- Stage 4/7: WMMA GEMM  y = scale * (X . Wq^T) + b -----------------------
// X: M x D f16 row-major; W: D x D f16 row-major (== column-major B operand).
// 64x128 block tile, BK=64, A+B staged to LDS via async DMA, double-buffered.
// mode 0: f16 out [B,H,S,HD]; mode 1: f16 out [B,H,HD,S]; mode 2: f32 out [M,D].
__global__ __launch_bounds__(256) void bitnet_gemm_kernel(
    const _Float16* __restrict__ X, const _Float16* __restrict__ W,
    const float* __restrict__ sumabs, const float* __restrict__ bias,
    _Float16* __restrict__ Yh, float* __restrict__ Yf, int mode) {
  using namespace bn;
  constexpr int LDA = 72;                      // 64 + 8 f16 pad (16B) vs bank conflicts
  __shared__ __align__(16) _Float16 At[2][64 * LDA];    // 2 x 9.0 KB
  __shared__ __align__(16) _Float16 Bt[2][128 * LDA];   // 2 x 18.0 KB
  const int tid = threadIdx.x;
  const int lane = tid & 31, wave = tid >> 5;
  const int row0 = blockIdx.y * 64;
  const int col0 = blockIdx.x * 128;
  const int wrow = (wave & 3) * 16;            // wave tile inside block
  const int wcol = (wave >> 2) * 64;

  // post one stage of async DMA: 2 A-chunks + 4 B-chunks of 16B per thread
  auto stage = [&](int buf, int k0) {
#pragma unroll
    for (int t = 0; t < 2; ++t) {
      const int c = tid + t * 256;             // 512 chunks cover 64x64 A tile
      const int r = c >> 3, cc = (c & 7) * 8;
      async_copy16(X + (size_t)(row0 + r) * D + k0 + cc, &At[buf][r * LDA + cc]);
    }
#pragma unroll
    for (int t = 0; t < 4; ++t) {
      const int c = tid + t * 256;             // 1024 chunks cover 128x64 B tile
      const int r = c >> 3, cc = (c & 7) * 8;
      async_copy16(W + (size_t)(col0 + r) * D + k0 + cc, &Bt[buf][r * LDA + cc]);
    }
  };

  v8f acc[4] = {};
  stage(0, 0);
  for (int k0 = 0; k0 < D; k0 += 64) {
    const int buf = (k0 >> 6) & 1;
    if (k0 + 64 < D) {
      stage(buf ^ 1, k0 + 64);                 // prefetch next stage into other buffer
      asm volatile("s_wait_asynccnt 0x6" ::: "memory");  // retire previous stage only
    } else {
      asm volatile("s_wait_asynccnt 0x0" ::: "memory");
    }
    __syncthreads();                           // stage visible to all waves
#pragma unroll
    for (int kk = 0; kk < 2; ++kk) {
      v16h a = load_frag_a(&At[buf][wrow * LDA], LDA, kk * 32, lane);
#pragma unroll
      for (int t = 0; t < 4; ++t) {
        v16h b = load_frag_b(&Bt[buf][(wcol + t * 16) * LDA], LDA, kk * 32, lane);
        acc[t] = WMMA_F16(a, b, acc[t]);
      }
    }
    __syncthreads();                           // done reading before DMA overwrites
  }

  const float scl = fminf(fmaxf(sumabs[0] * (1.0f / (float)(DD)), 1e-5f), 1000.0f);
  const int nn = lane & 15, hf = lane >> 4;
#pragma unroll
  for (int t = 0; t < 4; ++t) {
    const int col = col0 + wcol + t * 16 + nn;
    const float bv = bias[col];
    const int hh = col >> 7, dd = col & (HD - 1);
#pragma unroll
    for (int r = 0; r < 8; ++r) {
      const int row = row0 + wrow + 8 * hf + r;
      const float val = scl * acc[t][r] + bv;
      const int bb = row >> 11, ss = row & (S - 1);
      if (mode == 0)
        Yh[(((size_t)(bb * H + hh)) * S + ss) * HD + dd] = (_Float16)val;
      else if (mode == 1)
        Yh[(((size_t)(bb * H + hh)) * HD + dd) * S + ss] = (_Float16)val;
      else
        Yf[(size_t)row * D + col] = val;
    }
  }
}

// --- Stage 5: softmax stats (row max, sum of exp) per (b,h,q) ---------------
__global__ __launch_bounds__(256) void attn_stats_kernel(
    const _Float16* __restrict__ Qh, const _Float16* __restrict__ Kh,
    const int* __restrict__ mask, float* __restrict__ mrow, float* __restrict__ lrow) {
  using namespace bn;
  const int lane = threadIdx.x & 31, wave = threadIdx.x >> 5;
  const int bh = blockIdx.y, b = bh >> 4;
  const int q0 = blockIdx.x * 128 + wave * 16;
  const _Float16* Qp = Qh + ((size_t)bh * S + q0) * HD;
  const _Float16* Kp = Kh + (size_t)bh * S * HD;
  const int nn = lane & 15;

  v16h qa[4];
#pragma unroll
  for (int kk = 0; kk < 4; ++kk) qa[kk] = load_frag_a(Qp, HD, kk * 32, lane);

  v8f rm, rl;
#pragma unroll
  for (int i = 0; i < 8; ++i) { rm[i] = -1e30f; rl[i] = 0.f; }

  for (int kt = 0; kt < S; kt += 16) {
    v8f sc = {};
#pragma unroll
    for (int kk = 0; kk < 4; ++kk) {
      v16h kb = load_frag_b(Kp + (size_t)kt * HD, HD, kk * 32, lane);
      sc = WMMA_F16(qa[kk], kb, sc);
    }
    const float madd = (mask[b * S + kt + nn] != 0) ? 0.f : -1e30f;
    v8f s, tmax;
#pragma unroll
    for (int i = 0; i < 8; ++i) { s[i] = sc[i] * RSCALE + madd; tmax[i] = s[i]; }
#pragma unroll
    for (int d = 1; d < 16; d <<= 1)
#pragma unroll
      for (int i = 0; i < 8; ++i) tmax[i] = fmaxf(tmax[i], __shfl_xor(tmax[i], d, 32));
    v8f e;
#pragma unroll
    for (int i = 0; i < 8; ++i) {
      const float mn = fmaxf(rm[i], tmax[i]);
      e[i]  = __expf(s[i] - mn);
      rl[i] = rl[i] * __expf(rm[i] - mn);
      rm[i] = mn;
    }
#pragma unroll
    for (int d = 1; d < 16; d <<= 1)
#pragma unroll
      for (int i = 0; i < 8; ++i) e[i] += __shfl_xor(e[i], d, 32);
#pragma unroll
    for (int i = 0; i < 8; ++i) rl[i] += e[i];
  }
  if (nn == 0) {
    const int hf = lane >> 4;
#pragma unroll
    for (int r = 0; r < 8; ++r) {
      mrow[(size_t)bh * S + q0 + 8 * hf + r] = rm[r];
      lrow[(size_t)bh * S + q0 + 8 * hf + r] = rl[r];
    }
  }
}

// --- Stage 6: P = softmax(QK^T), attn_mean += P/H, O = P.V ------------------
__global__ __launch_bounds__(256) void attn_pv_kernel(
    const _Float16* __restrict__ Qh, const _Float16* __restrict__ Kh,
    const _Float16* __restrict__ Vt, const int* __restrict__ mask,
    const float* __restrict__ mrow, const float* __restrict__ lrow,
    float* __restrict__ attn_mean, _Float16* __restrict__ att16) {
  using namespace bn;
  __shared__ __align__(16) _Float16 ptile[8][16 * 32];   // per-wave 16x32 P tile
  const int lane = threadIdx.x & 31, wave = threadIdx.x >> 5;
  const int nn = lane & 15, hf = lane >> 4;
  const int bh = blockIdx.y, b = bh >> 4, h = bh & 15;
  const int q0 = blockIdx.x * 128 + wave * 16;
  const _Float16* Qp = Qh + ((size_t)bh * S + q0) * HD;
  const _Float16* Kp = Kh + (size_t)bh * S * HD;
  const _Float16* Vp = Vt + (size_t)bh * HD * S;

  v16h qa[4];
#pragma unroll
  for (int kk = 0; kk < 4; ++kk) qa[kk] = load_frag_a(Qp, HD, kk * 32, lane);

  float mr[8], li[8];
#pragma unroll
  for (int r = 0; r < 8; ++r) {
    mr[r] = mrow[(size_t)bh * S + q0 + 8 * hf + r];
    li[r] = 1.0f / lrow[(size_t)bh * S + q0 + 8 * hf + r];
  }
  v8f o[8] = {};
  _Float16* pt = &ptile[wave][0];

  for (int kt = 0; kt < S; kt += 32) {
    v8f s0 = {}, s1 = {};
#pragma unroll
    for (int kk = 0; kk < 4; ++kk) {
      v16h b0 = load_frag_b(Kp + (size_t)kt * HD, HD, kk * 32, lane);
      s0 = WMMA_F16(qa[kk], b0, s0);
      v16h b1 = load_frag_b(Kp + (size_t)(kt + 16) * HD, HD, kk * 32, lane);
      s1 = WMMA_F16(qa[kk], b1, s1);
    }
    const bool on0 = mask[b * S + kt + nn] != 0;
    const bool on1 = mask[b * S + kt + 16 + nn] != 0;
    float p0[8], p1[8];
#pragma unroll
    for (int r = 0; r < 8; ++r) {
      p0[r] = on0 ? __expf(s0[r] * RSCALE - mr[r]) * li[r] : 0.f;
      p1[r] = on1 ? __expf(s1[r] * RSCALE - mr[r]) * li[r] : 0.f;
    }
#pragma unroll
    for (int r = 0; r < 8; ++r) {
      const size_t rowb = ((size_t)b * S + q0 + 8 * hf + r) * S;
      atomicAdd(&attn_mean[rowb + kt + nn],      p0[r] * 0.0625f);
      atomicAdd(&attn_mean[rowb + kt + 16 + nn], p1[r] * 0.0625f);
    }
    // re-stripe P through LDS into A-fragment layout
#pragma unroll
    for (int r = 0; r < 8; ++r) {
      pt[(8 * hf + r) * 32 + nn]      = (_Float16)p0[r];
      pt[(8 * hf + r) * 32 + nn + 16] = (_Float16)p1[r];
    }
    asm volatile("s_wait_dscnt 0" ::: "memory");   // CDNA5 split DS counter
    union { v16h v; float4 f[2]; } u;
    u.f[0] = *(const float4*)&pt[nn * 32 + hf * 8];
    u.f[1] = *(const float4*)&pt[nn * 32 + hf * 8 + 16];
    const v16h pa = u.v;
#pragma unroll
    for (int t = 0; t < 8; ++t) {
      v16h vb = load_frag_b(Vp + (size_t)(t * 16) * S, S, kt, lane);
      o[t] = WMMA_F16(pa, vb, o[t]);
    }
  }
#pragma unroll
  for (int t = 0; t < 8; ++t)
#pragma unroll
    for (int r = 0; r < 8; ++r)
      att16[((size_t)b * S + q0 + 8 * hf + r) * D + h * HD + t * 16 + nn] =
          (_Float16)o[t][r];
}

// ---------------------------------------------------------------------------
extern "C" void kernel_launch(void* const* d_in, const int* in_sizes, int n_in,
                              void* d_out, int out_size, void* d_ws, size_t ws_size,
                              hipStream_t stream) {
  using namespace bn;
  const float* q_in = (const float*)d_in[0];
  const float* k_in = (const float*)d_in[1];
  const float* v_in = (const float*)d_in[2];
  const int*   mask = (const int*)d_in[3];
  const float* Wmat[4] = {(const float*)d_in[4], (const float*)d_in[5],
                          (const float*)d_in[6], (const float*)d_in[7]};
  const float* bq = (const float*)d_in[8];
  const float* bk = (const float*)d_in[9];
  const float* bv = (const float*)d_in[10];
  const float* bo = (const float*)d_in[11];

  char* ws = (char*)d_ws;
  float*    sumabs = (float*)(ws + OFS_SUM);
  _Float16* w16[4];
  for (int i = 0; i < 4; ++i) w16[i] = (_Float16*)(ws + OFS_W + (size_t)i * DD * 2);
  _Float16* x16   = (_Float16*)(ws + OFS_X16);   // reused later as attended f16
  _Float16* Qh    = (_Float16*)(ws + OFS_Q);
  _Float16* Kh    = (_Float16*)(ws + OFS_K);
  _Float16* Vt    = (_Float16*)(ws + OFS_VT);
  float*    mrow  = (float*)(ws + OFS_M);
  float*    lrow  = (float*)(ws + OFS_L);

  float* out       = (float*)d_out;              // [B,S,D]
  float* attn_mean = out + BSD;                  // [B,S,S]

  hipMemsetAsync(sumabs, 0, 4 * sizeof(float), stream);
  hipMemsetAsync(attn_mean, 0, (size_t)B * S * S * sizeof(float), stream);

  for (int i = 0; i < 4; ++i)
    abs_sum_kernel<<<dim3(256), dim3(256), 0, stream>>>(Wmat[i], sumabs + i, (int)DD);
  for (int i = 0; i < 4; ++i)
    quantize_kernel<<<dim3((unsigned)(DD / 256)), dim3(256), 0, stream>>>(
        Wmat[i], sumabs + i, w16[i], (int)DD);

  const dim3 ggrid(D / 128, M / 64), gblk(256);
  const dim3 agrid(S / 128, B * H);

  // Q projection
  convert_kernel<<<dim3((unsigned)(BSD / 256)), gblk, 0, stream>>>(q_in, x16, (int)BSD);
  bitnet_gemm_kernel<<<ggrid, gblk, 0, stream>>>(x16, w16[0], sumabs + 0, bq, Qh, nullptr, 0);
  // K projection
  convert_kernel<<<dim3((unsigned)(BSD / 256)), gblk, 0, stream>>>(k_in, x16, (int)BSD);
  bitnet_gemm_kernel<<<ggrid, gblk, 0, stream>>>(x16, w16[1], sumabs + 1, bk, Kh, nullptr, 0);
  // V projection (stored transposed per head: [B,H,HD,S])
  convert_kernel<<<dim3((unsigned)(BSD / 256)), gblk, 0, stream>>>(v_in, x16, (int)BSD);
  bitnet_gemm_kernel<<<ggrid, gblk, 0, stream>>>(x16, w16[2], sumabs + 2, bv, Vt, nullptr, 1);

  // attention
  attn_stats_kernel<<<agrid, gblk, 0, stream>>>(Qh, Kh, mask, mrow, lrow);
  attn_pv_kernel<<<agrid, gblk, 0, stream>>>(Qh, Kh, Vt, mask, mrow, lrow, attn_mean, x16);

  // output projection -> fp32 d_out
  bitnet_gemm_kernel<<<ggrid, gblk, 0, stream>>>(x16, w16[3], sumabs + 3, bo, nullptr, out, 2);
}
